// MultiHeadAttention_6107443495356
// MI455X (gfx1250) — matrix operand order repaired and checked by reading, not compiled, not run on hardware
//
#include <hip/hip_runtime.h>
#include <hip/hip_bf16.h>
#include <math.h>

// MI455X / gfx1250, wave32. Fused bf16-WMMA attention-score + softmax with
// async-LDS DMA staging of the `short` strip.
#define B_  2
#define S_  2048
#define D_  768
#define H_  12
#define DK_ 64

typedef __attribute__((ext_vector_type(16))) __bf16 v16bf;
typedef __attribute__((ext_vector_type(8)))  __bf16 v8bf;
typedef __attribute__((ext_vector_type(8)))  float  v8f;

// ---------------------------------------------------------------------------
// Kernel 1: W [k][n] f32  ->  WT [n][k] bf16   (so B-fragments load contiguous K)
// ---------------------------------------------------------------------------
__global__ __launch_bounds__(256) void wT_bf16_kernel(const float* __restrict__ W,
                                                      __bf16* __restrict__ WT) {
  int idx = blockIdx.x * 256 + threadIdx.x;       // over D*D outputs
  if (idx >= D_ * D_) return;
  int n = idx / D_;
  int k = idx - n * D_;
  WT[idx] = (__bf16)W[k * D_ + n];
}

// ---------------------------------------------------------------------------
// Kernel 2: Y[b,h,s,dk] = bf16( X[B*S, D] @ W[D, D] + bias )
// One wave -> one 16x16 tile; 8 waves/block cover 128 columns.
// A side: f32 loads + inline cvt to bf16. B side: contiguous bf16 from WT.
// ---------------------------------------------------------------------------
__global__ __launch_bounds__(256) void proj_gemm_kernel(const float* __restrict__ X,
                                                        const __bf16* __restrict__ WT,
                                                        const float* __restrict__ bias,
                                                        __bf16* __restrict__ Y) {
  const int lane = threadIdx.x & 31;
  const int wave = threadIdx.x >> 5;
  const int m0 = blockIdx.x * 16;                       // rows of B*S
  const int n0 = (blockIdx.y * 8 + wave) * 16;          // cols of D

  // A fragment addressing (16x32 bf16): lane[3:0] = M row, lane[4] selects K half.
  const int rowA = lane & 15;
  const int kbA  = (lane >> 4) << 4;                    // 0 or 16
  // B fragment addressing (32x16 bf16): lane[3:0] = N col, lane[4] selects K half.
  const int colB = lane & 15;
  const int khB  = (lane >> 4) << 4;                    // 0 or 16

  const float*  arow = X  + (size_t)(m0 + rowA) * D_;
  const __bf16* brow = WT + (size_t)(n0 + colB) * D_;

  v8f acc = {};
  for (int kc = 0; kc < D_; kc += 32) {
    v8f a0 = *(const v8f*)(arow + kc + kbA);
    v8f a1 = *(const v8f*)(arow + kc + kbA + 8);
    v16bf afrag;
#pragma unroll
    for (int i = 0; i < 8; ++i) {
      afrag[i]     = (__bf16)a0[i];
      afrag[8 + i] = (__bf16)a1[i];
    }
    v16bf bfrag = *(const v16bf*)(brow + kc + khB);
    acc = __builtin_amdgcn_wmma_f32_16x16x32_bf16(false, afrag, false, bfrag,
                                                  (short)0, acc, false, false);
  }

  // C/D layout: lane[3:0] = N, VGPR r = M row (+8 for lanes 16..31).
  const int gn = n0 + (lane & 15);
  const float bv = bias[gn];
  const int h = gn >> 6, dk = gn & 63;
#pragma unroll
  for (int r = 0; r < 8; ++r) {
    int m  = r + ((lane >> 4) << 3);
    int gm = m0 + m;
    int b  = gm >> 11;            // / S_
    int s  = gm & (S_ - 1);
    Y[(((size_t)(b * H_ + h)) * S_ + s) * DK_ + dk] = (__bf16)(acc[r] + bv);
  }
}

// ---------------------------------------------------------------------------
// Kernel 3: a = aspect @ Wd + bd ;  am[b,h,:] = a[b,:] @ weight_m[h]   (tiny)
// ---------------------------------------------------------------------------
__global__ __launch_bounds__(256) void aspect_am_kernel(const float* __restrict__ aspect,
                                                        const float* __restrict__ Wd,
                                                        const float* __restrict__ bd,
                                                        const float* __restrict__ weight_m,
                                                        float* __restrict__ am) {
  __shared__ float aS[B_ * DK_];
  int tid = threadIdx.x;
  if (tid < B_ * DK_) {
    int b = tid >> 6, e = tid & 63;
    float acc = bd[e];
    for (int d = 0; d < D_; ++d) acc += aspect[b * D_ + d] * Wd[d * DK_ + e];
    aS[tid] = acc;
  }
  __syncthreads();
  for (int idx = tid; idx < B_ * H_ * DK_; idx += 256) {
    int b = idx / (H_ * DK_);
    int rem = idx - b * (H_ * DK_);
    int h = rem >> 6, e = rem & 63;
    float acc = 0.f;
    for (int d2 = 0; d2 < DK_; ++d2)
      acc += aS[b * DK_ + d2] * weight_m[(h * DK_ + d2) * DK_ + e];
    am[idx] = acc;
  }
}

// ---------------------------------------------------------------------------
// Kernel 4: asp[b,h,t] = tanh( am[b,h,:] . K[b,h,t,:] + bias_m )
// ---------------------------------------------------------------------------
__global__ __launch_bounds__(256) void aspect_scores_kernel(const float* __restrict__ am,
                                                            const __bf16* __restrict__ Kbf,
                                                            const float* __restrict__ bias_m,
                                                            float* __restrict__ asp) {
  int idx = blockIdx.x * 256 + threadIdx.x;  // over B*H*S
  int bh = idx >> 11;
  int t  = idx & (S_ - 1);
  const __bf16* kr = Kbf + ((size_t)bh * S_ + t) * DK_;
  const float*  ar = am + bh * DK_;
  float acc = bias_m[0];
#pragma unroll 8
  for (int e = 0; e < DK_; ++e) acc += ar[e] * (float)kr[e];
  asp[idx] = tanhf(acc);
}

// ---------------------------------------------------------------------------
// Kernel 5 (fused, bandwidth-critical): one block per (b, h, 16 query rows).
// Step 1: async-DMA the contiguous 128 KB `short` strip straight into LDS
//         (global_load_async_to_lds_b128, tracked by ASYNCcnt).
// Step 2: WMMA Q.K^T tiles; epilogue does ds_add_f32 of the masked score
//         (+aspect broadcast) into the pre-loaded strip.
// Step 3: row softmax in LDS, single coalesced store of the output.
// ---------------------------------------------------------------------------
__global__ __launch_bounds__(256) void scores_softmax_kernel(const __bf16* __restrict__ Qbf,
                                                             const __bf16* __restrict__ Kbf,
                                                             const float* __restrict__ asp,
                                                             const int* __restrict__ mask,
                                                             const float* __restrict__ shrt,
                                                             float* __restrict__ out) {
  __shared__ float smem[16 * S_];       // 128 KB score strip (init = short)
  __shared__ float red[256];
  __shared__ float rowmax[16];
  __shared__ float rowrcp[16];

  const int lane = threadIdx.x & 31;
  const int wave = threadIdx.x >> 5;
  const int m0 = blockIdx.x * 16;       // query rows
  const int h  = blockIdx.y;
  const int b  = blockIdx.z;
  const int bh = b * H_ + h;

  // --- Async DMA: short[b,h,m0:m0+16,:] (contiguous 32768 f32) -> smem -----
  {
    const float* gsrc = shrt + ((size_t)bh * S_ + m0) * S_;
    const unsigned lbase = (unsigned)(size_t)&smem[0];
    const int tid = threadIdx.x;
#pragma unroll 4
    for (int rr = 0; rr < 32; ++rr) {           // 256 lanes x 16B per round
      const int e = rr * 1024 + tid * 4;        // float index
      unsigned laddr = lbase + (unsigned)e * 4u;
      unsigned long long gaddr = (unsigned long long)(size_t)(gsrc + e);
      asm volatile("global_load_async_to_lds_b128 %0, %1, off"
                   :: "v"(laddr), "v"(gaddr)
                   : "memory");
    }
  }

  const __bf16* qb = Qbf + (size_t)bh * S_ * DK_;
  const __bf16* kb = Kbf + (size_t)bh * S_ * DK_;

  // Load the A (Q-tile) fragments once: 16x64 = two 16x32 bf16 chunks.
  // (Overlaps with the in-flight async DMA.)
  const int rowA = lane & 15;
  const int kbA  = (lane >> 4) << 4;
  const __bf16* arow = qb + (size_t)(m0 + rowA) * DK_;
  v16bf afrag[2];
#pragma unroll
  for (int c = 0; c < 2; ++c) {
    v8bf lo = *(const v8bf*)(arow + c * 32 + kbA);
    v8bf hi = *(const v8bf*)(arow + c * 32 + kbA + 8);
#pragma unroll
    for (int i = 0; i < 8; ++i) {
      afrag[c][i]     = lo[i];
      afrag[c][8 + i] = hi[i];
    }
  }

  // Wait for this wave's async writes, then make all waves' writes visible.
  asm volatile("s_wait_asynccnt 0" ::: "memory");
  __syncthreads();

  const int colB = lane & 15;
  const int khB  = (lane >> 4) << 4;

  // 128 column tiles of 16; each of the 8 waves computes 16 tiles.
  for (int jj = 0; jj < 16; ++jj) {
    const int n0 = (wave * 16 + jj) * 16;
    const __bf16* brow = kb + (size_t)(n0 + colB) * DK_;
    v16bf b0 = *(const v16bf*)(brow + khB);
    v16bf b1 = *(const v16bf*)(brow + 32 + khB);

    v8f acc = {};
    acc = __builtin_amdgcn_wmma_f32_16x16x32_bf16(false, afrag[0], false, b0,
                                                  (short)0, acc, false, false);
    acc = __builtin_amdgcn_wmma_f32_16x16x32_bf16(false, afrag[1], false, b1,
                                                  (short)0, acc, false, false);

    const int t = n0 + (lane & 15);
    const float aspv = asp[(size_t)bh * S_ + t];
#pragma unroll
    for (int r = 0; r < 8; ++r) {
      int m  = r + ((lane >> 4) << 3);
      int sg = m0 + m;
      float val = acc[r] * 0.125f + aspv;                       // 1/sqrt(64)
      int mv = mask[((size_t)b * S_ + sg) * S_ + t];
      val = (mv == 0) ? -1e9f : val;
      // smem already holds `short`; reference does where(mask,..) + short.
      atomicAdd(&smem[m * S_ + t], val);                        // ds_add_f32
    }
  }
  __syncthreads();

  // Row max: 16 threads per row, strided scan.
  {
    int row = threadIdx.x >> 4, c = threadIdx.x & 15;
    float mx = -INFINITY;
    for (int t = c; t < S_; t += 16) mx = fmaxf(mx, smem[row * S_ + t]);
    red[threadIdx.x] = mx;
  }
  __syncthreads();
  if (threadIdx.x < 16) {
    float mx = -INFINITY;
    for (int j = 0; j < 16; ++j) mx = fmaxf(mx, red[threadIdx.x * 16 + j]);
    rowmax[threadIdx.x] = mx;
  }
  __syncthreads();
  // Row sum of exp.
  {
    int row = threadIdx.x >> 4, c = threadIdx.x & 15;
    float mx = rowmax[row], sum = 0.f;
    for (int t = c; t < S_; t += 16) sum += __expf(smem[row * S_ + t] - mx);
    red[threadIdx.x] = sum;
  }
  __syncthreads();
  if (threadIdx.x < 16) {
    float sum = 0.f;
    for (int j = 0; j < 16; ++j) sum += red[threadIdx.x * 16 + j];
    rowrcp[threadIdx.x] = 1.0f / sum;
  }
  __syncthreads();
  // Normalize + coalesced store.
  for (int r = 0; r < 16; ++r) {
    const int sg = m0 + r;
    const size_t base = ((size_t)bh * S_ + sg) * S_;
    const float mx = rowmax[r], rc = rowrcp[r];
    for (int t = threadIdx.x; t < S_; t += 256)
      out[base + t] = __expf(smem[r * S_ + t] - mx) * rc;
  }
}

// ---------------------------------------------------------------------------
extern "C" void kernel_launch(void* const* d_in, const int* in_sizes, int n_in,
                              void* d_out, int out_size, void* d_ws, size_t ws_size,
                              hipStream_t stream) {
  (void)in_sizes; (void)n_in; (void)out_size; (void)ws_size;

  const float* query    = (const float*)d_in[0];
  const float* key      = (const float*)d_in[1];
  const int*   mask     = (const int*)d_in[2];
  const float* shrt     = (const float*)d_in[3];
  const float* aspect   = (const float*)d_in[4];
  const float* Wq       = (const float*)d_in[5];
  const float* bq       = (const float*)d_in[6];
  const float* Wk       = (const float*)d_in[7];
  const float* bk       = (const float*)d_in[8];
  const float* Wd       = (const float*)d_in[9];
  const float* bd       = (const float*)d_in[10];
  const float* weight_m = (const float*)d_in[11];
  const float* bias_m   = (const float*)d_in[12];
  float* out = (float*)d_out;

  // Workspace carve-up (~15.2 MB total), 256B aligned.
  char* ws = (char*)d_ws;
  size_t off = 0;
  auto carve = [&](size_t bytes) -> void* {
    void* p = ws + off;
    off = (off + bytes + 255) & ~(size_t)255;
    return p;
  };
  __bf16* WqT = (__bf16*)carve((size_t)D_ * D_ * sizeof(__bf16));
  __bf16* WkT = (__bf16*)carve((size_t)D_ * D_ * sizeof(__bf16));
  __bf16* Qbf = (__bf16*)carve((size_t)B_ * H_ * S_ * DK_ * sizeof(__bf16));
  __bf16* Kbf = (__bf16*)carve((size_t)B_ * H_ * S_ * DK_ * sizeof(__bf16));
  float*  am  = (float*)carve((size_t)B_ * H_ * DK_ * sizeof(float));
  float*  asp = (float*)carve((size_t)B_ * H_ * S_ * sizeof(float));

  // 1) Weight transposes to bf16.
  wT_bf16_kernel<<<(D_ * D_ + 255) / 256, 256, 0, stream>>>(Wq, WqT);
  wT_bf16_kernel<<<(D_ * D_ + 255) / 256, 256, 0, stream>>>(Wk, WkT);

  // 2) Q / K projections: grid = (M tiles, N tile-groups of 8 waves).
  dim3 pgrid((B_ * S_) / 16, D_ / (16 * 8));
  proj_gemm_kernel<<<pgrid, 256, 0, stream>>>(query, WqT, bq, Qbf);
  proj_gemm_kernel<<<pgrid, 256, 0, stream>>>(key,   WkT, bk, Kbf);

  // 3) aspect -> am (tiny, single block).
  aspect_am_kernel<<<1, 256, 0, stream>>>(aspect, Wd, bd, weight_m, am);

  // 4) asp[b,h,t] = tanh(am.k + bias_m).
  aspect_scores_kernel<<<(B_ * H_ * S_) / 256, 256, 0, stream>>>(am, Kbf, bias_m, asp);

  // 5) Fused scores + mask + short(+async DMA) + softmax.
  dim3 sgrid(S_ / 16, H_, B_);
  scores_softmax_kernel<<<sgrid, 256, 0, stream>>>(Qbf, Kbf, asp, mask, shrt, out);
}
